// RecurrentRGCN_26422638805226
// MI455X (gfx1250) — matrix operand order repaired and checked by reading, not compiled, not run on hardware
//
#include <hip/hip_runtime.h>
#include <math.h>

// ---------------- problem constants (match reference) ----------------
#define NN   200000   // entities
#define R2   480      // relations*2
#define HD   128      // hidden dim
#define TT   3        // timesteps
#define EE   400000   // edges per snapshot
#define R2E  600000   // r_to_e entries per snapshot
#define LL   2        // rgcn layers
#define LEAK 0.22916667f

typedef __bf16 v16bf __attribute__((ext_vector_type(16)));
typedef float  v8f   __attribute__((ext_vector_type(8)));
typedef unsigned int u32x4 __attribute__((ext_vector_type(4)));
typedef int          i32x4 __attribute__((ext_vector_type(4)));
typedef int          i32x8 __attribute__((ext_vector_type(8)));

// =====================================================================
// Tensor Data Mover: 2D tile load (rows x cols bf16, contiguous) from
// global memory into LDS.  D# descriptor per ISA 08 ch.8:
//  group0: count=1 | lds_addr | global_addr(57b) | type=2
//  group1: data_size=1(2B) | tensor_dim0/1 | tile_dim0/1 | dim0_stride
// Issued by one wave; completion via s_wait_tensorcnt.
// =====================================================================
__device__ __forceinline__ void tdm_load_2d_bf16(const void* gptr, unsigned lds_off,
                                                 int rows, int cols) {
#if __has_builtin(__builtin_amdgcn_tensor_load_to_lds)
    const unsigned long long ga = (unsigned long long)gptr;
    const u32x4 g0 = {
        1u,                                                   // count=1 (valid), user mode
        lds_off,                                              // lds_addr (bytes)
        (unsigned)(ga & 0xFFFFFFFFu),                         // global_addr[31:0]
        (unsigned)((ga >> 32) & 0x1FFFFFFu) | (2u << 30)      // global_addr[56:32] | type=2
    };
    const i32x8 g1 = {
        (int)(1u << 16),                                      // data_size=1 -> 2 bytes/elem
        (int)((cols & 0xFFFF) << 16),                         // tensor_dim0[15:0] @ bits63:48
        (int)(((cols >> 16) & 0xFFFF) | ((rows & 0xFFFF) << 16)), // dim0 hi | tensor_dim1 lo
        (int)(((rows >> 16) & 0xFFFF) | ((cols & 0xFFFF) << 16)), // dim1 hi | tile_dim0
        (int)(rows & 0xFFFF),                                 // tile_dim1 (tile_dim2=0)
        cols,                                                 // tensor_dim0_stride[31:0]
        0, 0                                                  // stride hi, tensor_dim1_stride
    };
    const i32x4 z4 = {0, 0, 0, 0};
#if defined(__clang_major__) && (__clang_major__ >= 23)
    const i32x8 z8 = {0, 0, 0, 0, 0, 0, 0, 0};
    __builtin_amdgcn_tensor_load_to_lds(g0, g1, z4, z4, z8, 0);
#else
    __builtin_amdgcn_tensor_load_to_lds(g0, g1, z4, z4, 0);
#endif
#endif
}

__device__ __forceinline__ void tdm_wait0() {
#if __has_builtin(__builtin_amdgcn_s_wait_tensorcnt)
    __builtin_amdgcn_s_wait_tensorcnt(0);
#endif
}

// low 32 bits of a generic LDS pointer = LDS byte offset (ISA §10.2 aperture rules)
__device__ __forceinline__ unsigned lds_offset_of(const void* p) {
    return (unsigned)(unsigned long long)p;
}

// =====================================================================
// Convert the three weight stacks (L x 128 x 128 each) to bf16 shadows.
// =====================================================================
__global__ void k_cvt_weights(const float* __restrict__ a, const float* __restrict__ b,
                              const float* __restrict__ c,
                              __bf16* __restrict__ ab, __bf16* __restrict__ bb,
                              __bf16* __restrict__ cb, int n) {
    int i = blockIdx.x * blockDim.x + threadIdx.x;
    if (i < n) {
        ab[i] = (__bf16)a[i];
        bb[i] = (__bf16)b[i];
        cb[i] = (__bf16)c[i];
    }
}

// =====================================================================
// Row-wise L2 normalize (wave32-aware): out_f32 = x/||x||, bf16 shadow,
// optional extra f32 copy (history output). One wave per 128-dim row.
// =====================================================================
__global__ void k_norm_rows(const float* __restrict__ in, float* __restrict__ out_f32,
                            __bf16* __restrict__ out_bf, float* __restrict__ out_copy,
                            int rows) {
    const int wave = threadIdx.x >> 5;
    const int lane = threadIdx.x & 31;
    const int row  = blockIdx.x * 8 + wave;
    if (row >= rows) return;
    const size_t base = (size_t)row * HD;
    float v[4];
    float ss = 0.f;
#pragma unroll
    for (int i = 0; i < 4; ++i) {
        v[i] = in[base + lane + 32 * i];
        ss += v[i] * v[i];
    }
#pragma unroll
    for (int m = 16; m >= 1; m >>= 1) ss += __shfl_xor(ss, m, 32);
    const float inv = 1.0f / fmaxf(sqrtf(ss), 1e-12f);
#pragma unroll
    for (int i = 0; i < 4; ++i) {
        const float o = v[i] * inv;
        out_f32[base + lane + 32 * i] = o;
        out_bf [base + lane + 32 * i] = (__bf16)o;
        if (out_copy) out_copy[base + lane + 32 * i] = o;
    }
}

// =====================================================================
// Relation pool: segment sum+count privatized in a 248KB LDS pool per
// workgroup (CDNA5 WGP has 320KB LDS), one WAVE per r_to_e entry:
// lane handles 4 contiguous columns -> coalesced 8B gathers + spread
// ds_add_f32 banks.  Flushed once with global atomics.
// dyn LDS = (R2*HD + R2) floats.
// =====================================================================
__global__ void k_pool(const __bf16* __restrict__ h_bf, const int* __restrict__ rte,
                       const int* __restrict__ rel, float* __restrict__ pool_sum,
                       float* __restrict__ pool_cnt) {
    extern __shared__ float sm[];
    float* ps = sm;             // [R2*HD]
    float* pc = sm + R2 * HD;   // [R2]
    const int tid = threadIdx.x;
    for (int i = tid; i < R2 * HD + R2; i += blockDim.x) sm[i] = 0.f;
    __syncthreads();

    const int lane   = tid & 31;
    const int nwaves = (gridDim.x * blockDim.x) >> 5;
    const int wid    = (blockIdx.x * blockDim.x + tid) >> 5;
    const int c0     = lane * 4;
    for (int e = wid; e < R2E; e += nwaves) {
        const int node = rte[e];
        const int r    = rel[e];
        const __bf16* hr = h_bf + (size_t)node * HD + c0;
        float* dp = ps + r * HD + c0;
#pragma unroll
        for (int c = 0; c < 4; ++c) atomicAdd(&dp[c], (float)hr[c]);
        if (lane == 0) atomicAdd(&pc[r], 1.0f);
    }
    __syncthreads();
    for (int i = tid; i < R2 * HD; i += blockDim.x) atomicAdd(&pool_sum[i], ps[i]);
    for (int i = tid; i < R2; i += blockDim.x)      atomicAdd(&pool_cnt[i], pc[i]);
}

// =====================================================================
// GRU cell over 480 relation rows + L2 norm. One block (128 thr) / row.
// =====================================================================
__global__ void k_gru(const float* __restrict__ emb_rel, const float* __restrict__ pool_sum,
                      const float* __restrict__ pool_cnt,
                      const float* __restrict__ w_ih, const float* __restrict__ w_hh,
                      const float* __restrict__ b_ih, const float* __restrict__ b_hh,
                      float* __restrict__ h0, __bf16* __restrict__ h0_bf,
                      float* __restrict__ rel_out) {
    __shared__ float xin[2 * HD];
    __shared__ float hrow[HD];
    __shared__ float red[HD];
    const int r = blockIdx.x;
    const int j = threadIdx.x;              // 0..127
    const size_t rb = (size_t)r * HD;

    xin[j] = emb_rel[rb + j];
    const float cnt = pool_cnt[r];
    xin[HD + j] = (cnt > 0.f) ? pool_sum[rb + j] / fmaxf(cnt, 1.0f) : 0.f;
    hrow[j] = h0[rb + j];
    __syncthreads();

    float gi[3], gh[3];
#pragma unroll
    for (int g = 0; g < 3; ++g) {
        const int o = g * HD + j;
        float si = b_ih[o];
        const float* wi = w_ih + (size_t)o * (2 * HD);
        for (int k = 0; k < 2 * HD; ++k) si += wi[k] * xin[k];
        gi[g] = si;
        float sh = b_hh[o];
        const float* wh = w_hh + (size_t)o * HD;
        for (int k = 0; k < HD; ++k) sh += wh[k] * hrow[k];
        gh[g] = sh;
    }
    const float rg = 1.f / (1.f + expf(-(gi[0] + gh[0])));
    const float zg = 1.f / (1.f + expf(-(gi[1] + gh[1])));
    const float ng = tanhf(gi[2] + rg * gh[2]);
    const float hn = (1.f - zg) * ng + zg * hrow[j];

    red[j] = hn * hn;
    __syncthreads();
    for (int s = 64; s > 0; s >>= 1) {
        if (j < s) red[j] += red[j + s];
        __syncthreads();
    }
    const float inv = 1.f / fmaxf(sqrtf(red[0]), 1e-12f);
    const float o = hn * inv;
    h0[rb + j]     = o;
    h0_bf[rb + j]  = (__bf16)o;
    rel_out[rb + j]= o;
}

// =====================================================================
// Edge-message GEMM + scatter.  Tile = 16 edges, 256 thr = 8 waves,
// wave w owns output cols [16w,16w+16).  128x128 bf16 weight is DMA'd
// into LDS by the Tensor Data Mover; B-fragments stay in VGPRs; A comes
// from an LDS-staged tile of gathered (h[src]+rel[et]) rows.
// Ragged rows are routed to a dummy accumulator row (index NN) so the
// scatter atomics are UNCONDITIONAL -> no exec-mask churn in the loop.
// dyn LDS: HD*HD*2 (weight) + 16*HD*2 (x tile) + 64 (dst ids).
// =====================================================================
__global__ void k_edge(const __bf16* __restrict__ h_bf, const __bf16* __restrict__ h0_bf,
                       const int* __restrict__ src, const int* __restrict__ dst,
                       const int* __restrict__ et, const __bf16* __restrict__ wn_bf,
                       float* __restrict__ out /*[NN+1][HD]*/, float* __restrict__ deg /*[NN+1]*/,
                       int nedges, int computeDeg) {
    extern __shared__ char smem[];
    __bf16* wlds = (__bf16*)smem;                                   // [HD*HD]
    __bf16 (*xs)[HD] = (__bf16(*)[HD])(smem + HD * HD * 2);         // [16][HD]
    int* sdst = (int*)(smem + HD * HD * 2 + 16 * HD * 2);           // [16]

    const int tid  = threadIdx.x;
    const int wave = tid >> 5;              // 0..7 -> n-tile
    const int lane = tid & 31;
    const int ncol = wave * 16 + (lane & 15);
    const int kb_b = (lane >= 16) ? 16 : 0; // B: K split at 16
    const int kb_a = (lane >= 16) ? 8 : 0;  // unused: A also splits K at 16 via kb_b
    const int arow = lane & 15;
    const int roff = (lane >= 16) ? 8 : 0;

    // ---- stage weight into LDS (TDM) ----
#if __has_builtin(__builtin_amdgcn_tensor_load_to_lds)
    if (wave == 0) {
        tdm_load_2d_bf16(wn_bf, lds_offset_of(wlds), HD, HD);
        tdm_wait0();
    }
#else
    for (int i = tid; i < HD * HD; i += blockDim.x) wlds[i] = wn_bf[i];
#endif
    __syncthreads();

    // resident B fragments from LDS: lane=col, element e -> K = kb*32 + kb_b + e
    v16bf bfrag[4];
#pragma unroll
    for (int kb = 0; kb < 4; ++kb)
#pragma unroll
        for (int e = 0; e < 16; ++e)
            bfrag[kb][e] = wlds[(kb * 32 + kb_b + e) * HD + ncol];

    const int ntiles = (nedges + 15) / 16;
    for (int tile = blockIdx.x; tile < ntiles; tile += gridDim.x) {
        const int ebase = tile * 16;
        const bool fullTile = (ebase + 16 <= nedges);   // block-uniform
        // prefetch next tile's index data (global_prefetch_b8)
        const int pbase = (tile + gridDim.x) * 16;
        if (pbase < nedges && tid < 16) {
            __builtin_prefetch(&src[pbase + tid], 0, 1);
            __builtin_prefetch(&et [pbase + tid], 0, 1);
            __builtin_prefetch(&dst[pbase + tid], 0, 1);
        }
        {   // stage x = h[src]+rel[et] : thread -> (row=tid/16, 8 cols)
            const int row = tid >> 4;
            const int c0  = (tid & 15) * 8;
            const int eidx = ebase + row;
            if (fullTile || eidx < nedges) {
                const int s = src[eidx];
                const int r = et[eidx];
                const __bf16* hs = h_bf  + (size_t)s * HD;
                const __bf16* rs = h0_bf + (size_t)r * HD;
#pragma unroll
                for (int c = 0; c < 8; ++c)
                    xs[row][c0 + c] = (__bf16)((float)hs[c0 + c] + (float)rs[c0 + c]);
                if ((tid & 15) == 0) sdst[row] = dst[eidx];
            } else {
#pragma unroll
                for (int c = 0; c < 8; ++c) xs[row][c0 + c] = (__bf16)0.f;
                if ((tid & 15) == 0) sdst[row] = NN;   // dummy accumulator row
            }
        }
        __syncthreads();

        // cache this lane's 8 destination ids in VGPRs (2x ds_load_b128)
        int myd[8];
#pragma unroll
        for (int r = 0; r < 8; ++r) myd[r] = sdst[roff + r];

        if (computeDeg && tid < 16) atomicAdd(&deg[sdst[tid]], 1.0f);

        v8f acc = {};
#pragma unroll
        for (int kb = 0; kb < 4; ++kb) {
            v16bf afrag;
#pragma unroll
            for (int e = 0; e < 16; ++e)
                afrag[e] = xs[arow][kb * 32 + kb_b + e];
            acc = __builtin_amdgcn_wmma_f32_16x16x32_bf16(
                false, afrag, false, bfrag[kb], (short)0, acc, false, false);
        }

        // unconditional scatter-add: (vgpr r, lane) = row r+roff, col lane%16
#pragma unroll
        for (int r = 0; r < 8; ++r)
            atomicAdd(&out[(size_t)myd[r] * HD + ncol], acc[r]);
        __syncthreads();
    }
}

// =====================================================================
// Self-loop GEMM (both loop_w and evolve_w TDM'd to LDS, per-row select
// by deg) + combine with agg*inv_deg + leaky-ReLU.  Same WMMA structure
// as k_edge; writes f32 in-place over agg and a bf16 shadow.  Full-tile
// fast path (block-uniform) keeps the epilogue unpredicated.
// dyn LDS: 2*HD*HD*2 (weights) + 16*HD*2 (x tile).
// =====================================================================
__global__ void k_selfloop(const __bf16* __restrict__ h_in_bf, const __bf16* __restrict__ lw_bf,
                           const __bf16* __restrict__ ew_bf, const float* __restrict__ deg,
                           float* __restrict__ inout, __bf16* __restrict__ out_bf, int nnodes) {
    extern __shared__ char smem[];
    __bf16* wl = (__bf16*)smem;                                     // [HD*HD]
    __bf16* we = wl + HD * HD;                                      // [HD*HD]
    __bf16 (*xs)[HD] = (__bf16(*)[HD])(smem + 2 * HD * HD * 2);     // [16][HD]

    const int tid  = threadIdx.x;
    const int wave = tid >> 5;
    const int lane = tid & 31;
    const int ncol = wave * 16 + (lane & 15);
    const int kb_b = (lane >= 16) ? 16 : 0;
    const int kb_a = (lane >= 16) ? 8 : 0;
    const int arow = lane & 15;
    const int roff = (lane >= 16) ? 8 : 0;

#if __has_builtin(__builtin_amdgcn_tensor_load_to_lds)
    if (wave == 0) {
        tdm_load_2d_bf16(lw_bf, lds_offset_of(wl), HD, HD);
        tdm_load_2d_bf16(ew_bf, lds_offset_of(we), HD, HD);
        tdm_wait0();
    }
#else
    for (int i = tid; i < HD * HD; i += blockDim.x) { wl[i] = lw_bf[i]; we[i] = ew_bf[i]; }
#endif
    __syncthreads();

    v16bf bl[4], be[4];
#pragma unroll
    for (int kb = 0; kb < 4; ++kb)
#pragma unroll
        for (int e = 0; e < 16; ++e) {
            const int wi = (kb * 32 + kb_b + e) * HD + ncol;
            bl[kb][e] = wl[wi];
            be[kb][e] = we[wi];
        }

    const int ntiles = (nnodes + 15) / 16;
    for (int tile = blockIdx.x; tile < ntiles; tile += gridDim.x) {
        const int nbase = tile * 16;
        const bool fullTile = (nbase + 16 <= nnodes);   // block-uniform
        {
            const int row = tid >> 4;
            const int c0  = (tid & 15) * 8;
            const int nidx = nbase + row;
            if (fullTile || nidx < nnodes) {
                const __bf16* hs = h_in_bf + (size_t)nidx * HD;
#pragma unroll
                for (int c = 0; c < 8; ++c) xs[row][c0 + c] = hs[c0 + c];
            } else {
#pragma unroll
                for (int c = 0; c < 8; ++c) xs[row][c0 + c] = (__bf16)0.f;
            }
        }
        __syncthreads();

        v8f accL = {}, accE = {};
#pragma unroll
        for (int kb = 0; kb < 4; ++kb) {
            v16bf afrag;
#pragma unroll
            for (int e = 0; e < 16; ++e)
                afrag[e] = xs[arow][kb * 32 + kb_b + e];
            accL = __builtin_amdgcn_wmma_f32_16x16x32_bf16(
                false, afrag, false, bl[kb], (short)0, accL, false, false);
            accE = __builtin_amdgcn_wmma_f32_16x16x32_bf16(
                false, afrag, false, be[kb], (short)0, accE, false, false);
        }

        if (fullTile) {
#pragma unroll
            for (int r = 0; r < 8; ++r) {
                const int node = nbase + roff + r;
                const float dg = deg[node];
                const float loop = (dg > 0.f) ? accL[r] : accE[r];
                const float invd = (dg > 0.f) ? 1.0f / fmaxf(dg, 1.0f) : 0.0f;
                const size_t idx = (size_t)node * HD + ncol;
                float v = inout[idx] * invd + loop;
                v = (v >= 0.f) ? v : LEAK * v;
                inout[idx]  = v;
                out_bf[idx] = (__bf16)v;
            }
        } else {
#pragma unroll
            for (int r = 0; r < 8; ++r) {
                const int node = nbase + roff + r;
                if (node < nnodes) {
                    const float dg = deg[node];
                    const float loop = (dg > 0.f) ? accL[r] : accE[r];
                    const float invd = (dg > 0.f) ? 1.0f / fmaxf(dg, 1.0f) : 0.0f;
                    const size_t idx = (size_t)node * HD + ncol;
                    float v = inout[idx] * invd + loop;
                    v = (v >= 0.f) ? v : LEAK * v;
                    inout[idx]  = v;
                    out_bf[idx] = (__bf16)v;
                }
            }
        }
        __syncthreads();
    }
}

// =====================================================================
// Host-side orchestration
// =====================================================================
extern "C" void kernel_launch(void* const* d_in, const int* in_sizes, int n_in,
                              void* d_out, int out_size, void* d_ws, size_t ws_size,
                              hipStream_t stream) {
    const float* dynamic_emb = (const float*)d_in[0];
    const float* emb_rel     = (const float*)d_in[1];
    const float* w_ih        = (const float*)d_in[2];
    const float* w_hh        = (const float*)d_in[3];
    const float* b_ih        = (const float*)d_in[4];
    const float* b_hh        = (const float*)d_in[5];
    const float* w_neighbor  = (const float*)d_in[6];
    const float* loop_w      = (const float*)d_in[7];
    const float* evolve_w    = (const float*)d_in[8];
    const int*   src         = (const int*)d_in[9];
    const int*   dst         = (const int*)d_in[10];
    const int*   etype       = (const int*)d_in[11];
    const int*   r_to_e      = (const int*)d_in[12];
    const int*   rel_of_rte  = (const int*)d_in[13];

    float* out_hist = (float*)d_out;                              // [T,N,HD]
    float* out_rel  = (float*)d_out + (size_t)TT * NN * HD;       // [T,R2,HD]

    // ---- scratch layout (NN+1 rows: last row is the atomic dump row) ----
    char* w = (char*)d_ws;
    size_t off = 0;
    auto alloc = [&](size_t bytes) -> void* {
        void* p = w + off;
        off = (off + bytes + 255) & ~(size_t)255;
        return p;
    };
    float*  bufA_f32 = (float*) alloc((size_t)(NN + 1) * HD * 4);
    float*  bufB_f32 = (float*) alloc((size_t)(NN + 1) * HD * 4);
    float*  deg      = (float*) alloc((size_t)(NN + 1) * 4);
    float*  pool_sum = (float*) alloc((size_t)R2 * HD * 4);
    float*  pool_cnt = (float*) alloc((size_t)R2 * 4);
    float*  h0_f32   = (float*) alloc((size_t)R2 * HD * 4);
    __bf16* bufA_bf  = (__bf16*)alloc((size_t)NN * HD * 2);
    __bf16* bufB_bf  = (__bf16*)alloc((size_t)NN * HD * 2);
    __bf16* h0_bf    = (__bf16*)alloc((size_t)R2 * HD * 2);
    __bf16* wn_bf    = (__bf16*)alloc((size_t)LL * HD * HD * 2);
    __bf16* lw_bf    = (__bf16*)alloc((size_t)LL * HD * HD * 2);
    __bf16* ew_bf    = (__bf16*)alloc((size_t)LL * HD * HD * 2);
    (void)ws_size; (void)in_sizes; (void)n_in; (void)out_size;

    // ---- one-time prep ----
    {
        const int nw = LL * HD * HD;
        k_cvt_weights<<<(nw + 255) / 256, 256, 0, stream>>>(
            w_neighbor, loop_w, evolve_w, wn_bf, lw_bf, ew_bf, nw);
    }
    k_norm_rows<<<(NN + 7) / 8, 256, 0, stream>>>(dynamic_emb, bufA_f32, bufA_bf, nullptr, NN);
    hipMemcpyAsync(h0_f32, emb_rel, (size_t)R2 * HD * 4, hipMemcpyDeviceToDevice, stream);

    const size_t POOL_LDS = ((size_t)R2 * HD + R2) * sizeof(float);   // ~248 KB
    const size_t EDGE_LDS = (size_t)HD * HD * 2 + 16 * HD * 2 + 64;   // 36 KB + ids
    const size_t LOOP_LDS = (size_t)2 * HD * HD * 2 + 16 * HD * 2;    // 68 KB

    for (int t = 0; t < TT; ++t) {
        const int* rte_t = r_to_e     + (size_t)t * R2E;
        const int* rel_t = rel_of_rte + (size_t)t * R2E;
        const int* src_t = src        + (size_t)t * EE;
        const int* dst_t = dst        + (size_t)t * EE;
        const int* et_t  = etype      + (size_t)t * EE;

        // relation pool -> GRU -> h0
        hipMemsetAsync(pool_sum, 0, (size_t)R2 * HD * 4, stream);
        hipMemsetAsync(pool_cnt, 0, (size_t)R2 * 4, stream);
        k_pool<<<240, 256, POOL_LDS, stream>>>(bufA_bf, rte_t, rel_t, pool_sum, pool_cnt);
        k_gru<<<R2, HD, 0, stream>>>(emb_rel, pool_sum, pool_cnt, w_ih, w_hh, b_ih, b_hh,
                                     h0_f32, h0_bf, out_rel + (size_t)t * R2 * HD);

        hipMemsetAsync(deg, 0, (size_t)NN * 4, stream);

        // ---- RGCN layer 0: bufA -> bufB ----
        hipMemsetAsync(bufB_f32, 0, (size_t)NN * HD * 4, stream);
        k_edge<<<2048, 256, EDGE_LDS, stream>>>(bufA_bf, h0_bf, src_t, dst_t, et_t,
                                                wn_bf, bufB_f32, deg, EE, 1);
        k_selfloop<<<2048, 256, LOOP_LDS, stream>>>(bufA_bf, lw_bf, ew_bf,
                                                    deg, bufB_f32, bufB_bf, NN);

        // ---- RGCN layer 1: bufB -> bufA ----
        hipMemsetAsync(bufA_f32, 0, (size_t)NN * HD * 4, stream);
        k_edge<<<2048, 256, EDGE_LDS, stream>>>(bufB_bf, h0_bf, src_t, dst_t, et_t,
                                                wn_bf + (size_t)HD * HD, bufA_f32, deg, EE, 0);
        k_selfloop<<<2048, 256, LOOP_LDS, stream>>>(bufB_bf, lw_bf + (size_t)HD * HD,
                                                    ew_bf + (size_t)HD * HD,
                                                    deg, bufA_f32, bufA_bf, NN);

        // h = l2norm(cur); also emit history_embs[t]
        k_norm_rows<<<(NN + 7) / 8, 256, 0, stream>>>(
            bufA_f32, bufA_f32, bufA_bf, out_hist + (size_t)t * NN * HD, NN);
    }
}
